// GRU_Model_16106127360422
// MI455X (gfx1250) — matrix operand order repaired and checked by reading, not compiled, not run on hardware
//
#include <hip/hip_runtime.h>
#include <hip/hip_bf16.h>
#include <math.h>

// ---------------------------------------------------------------------------
// Problem constants (from reference): V=50000 E=300 H=512 B=128 T=256 O=8
// ---------------------------------------------------------------------------
#define BB   128
#define TT   256
#define HH   512
#define EE   300
#define EP   320          // E padded to a multiple of 64 for WMMA K pipeline
#define GG   1536         // 3*H (gate dimension)
#define MM   (BB*TT)      // 32768 rows for batched projections
#define LDST 66           // padded LDS row stride (floats): 8*66 % 64 == 16 -> no bank conflicts

typedef _Float16 v16h __attribute__((ext_vector_type(16)));
typedef _Float16 v8h  __attribute__((ext_vector_type(8)));
typedef float    v8f  __attribute__((ext_vector_type(8)));

// ---------------------------------------------------------------------------
// Wave-level WMMA GEMM: one wave computes a 16(M) x 64(N) accumulator tile of
//   acc = A[16,K] * W[64,K]^T + bias       (f16 inputs, f32 accumulate)
// A and W are row-major; W rows are the N dimension (PyTorch [N,K] layout),
// which is exactly the transposed-B layout WMMA wants -> contiguous 16B loads.
//
// Fragment mapping used here (wave32; A/B use contiguous k, same order in both):
//   A 16x32 f16 : lane l -> row (l&15); halves 0..15 = K (l>>4)*16 .. +15
//   B 32x16 f16 : lane l -> col (l&15); halves 0..15 = K (l>>4)*16 .. +15
//   C/D 16x16 f32: lane l -> col (l&15); vgpr j -> row (l>>4)*8 + j
//
// The K loop is software-pipelined with two alternating fragment sets so the
// 10 global_load_b128 of the NEXT k-slice are in flight while the current
// 4 v_wmma execute (K must be a multiple of 64; true for 320/512/1024).
// ---------------------------------------------------------------------------
__device__ __forceinline__ v16h cat8(v8h lo, v8h hi) {
  v16h r;
#pragma unroll
  for (int i = 0; i < 8; ++i) { r[i] = lo[i]; r[i + 8] = hi[i]; }
  return r;
}

struct Frags {
  v8h al, ah, b0l, b0h, b1l, b1h, b2l, b2h, b3l, b3h;
};

__device__ __forceinline__ void load_frags(
    Frags& f, const _Float16* __restrict__ arow,
    const _Float16* __restrict__ w0, const _Float16* __restrict__ w1,
    const _Float16* __restrict__ w2, const _Float16* __restrict__ w3, int kk)
{
  f.al  = *(const v8h*)(arow + kk);  f.ah  = *(const v8h*)(arow + kk + 8);
  f.b0l = *(const v8h*)(w0 + kk);    f.b0h = *(const v8h*)(w0 + kk + 8);
  f.b1l = *(const v8h*)(w1 + kk);    f.b1h = *(const v8h*)(w1 + kk + 8);
  f.b2l = *(const v8h*)(w2 + kk);    f.b2h = *(const v8h*)(w2 + kk + 8);
  f.b3l = *(const v8h*)(w3 + kk);    f.b3h = *(const v8h*)(w3 + kk + 8);
}

__device__ __forceinline__ void do_wmma(const Frags& f, v8f& c0, v8f& c1, v8f& c2, v8f& c3)
{
  v16h a = cat8(f.al, f.ah);
  c0 = __builtin_amdgcn_wmma_f32_16x16x32_f16(false, a, false, cat8(f.b0l, f.b0h), (short)0, c0, false, false);
  c1 = __builtin_amdgcn_wmma_f32_16x16x32_f16(false, a, false, cat8(f.b1l, f.b1h), (short)0, c1, false, false);
  c2 = __builtin_amdgcn_wmma_f32_16x16x32_f16(false, a, false, cat8(f.b2l, f.b2h), (short)0, c2, false, false);
  c3 = __builtin_amdgcn_wmma_f32_16x16x32_f16(false, a, false, cat8(f.b3l, f.b3h), (short)0, c3, false, false);
}

__device__ __forceinline__ void wave_gemm_16x64_acc(
    const _Float16* __restrict__ A, size_t lda,
    const _Float16* __restrict__ W, size_t ldw,
    const float* __restrict__ bias, int K,
    v8f& c0, v8f& c1, v8f& c2, v8f& c3)
{
  const int lane = threadIdx.x & 31;
  const int half = lane >> 4;   // 0/1
  const int l15  = lane & 15;

  const float b0 = bias[l15], b1 = bias[16 + l15], b2 = bias[32 + l15], b3 = bias[48 + l15];
#pragma unroll
  for (int j = 0; j < 8; ++j) { c0[j] = b0; c1[j] = b1; c2[j] = b2; c3[j] = b3; }

  const _Float16* arow = A + (size_t)l15 * lda + half * 16;
  const _Float16* w0 = W + (size_t)(l15     ) * ldw + half * 16;
  const _Float16* w1 = W + (size_t)(l15 + 16) * ldw + half * 16;
  const _Float16* w2 = W + (size_t)(l15 + 32) * ldw + half * 16;
  const _Float16* w3 = W + (size_t)(l15 + 48) * ldw + half * 16;

  Frags f0, f1;
  load_frags(f0, arow, w0, w1, w2, w3, 0);
#pragma unroll 1
  for (int kk = 0; kk < K; kk += 64) {
    load_frags(f1, arow, w0, w1, w2, w3, kk + 32);   // prefetch slice kk+32
    do_wmma(f0, c0, c1, c2, c3);                      // consume slice kk
    if (kk + 64 < K)
      load_frags(f0, arow, w0, w1, w2, w3, kk + 64);  // prefetch slice kk+64
    do_wmma(f1, c0, c1, c2, c3);                      // consume slice kk+32
  }
}

// ---------------------------------------------------------------------------
// Batched projection GEMM (f16 out). Block = 256 threads = 8 waves
// -> block tile 64(M) x 128(N). grid.x = N/128, grid.y = M/64.
// ---------------------------------------------------------------------------
__global__ __launch_bounds__(256) void gemm_f16_kernel(
    const _Float16* __restrict__ A, int lda,
    const _Float16* __restrict__ W, int ldw,
    const float* __restrict__ bias,
    _Float16* __restrict__ out, int ldo, int K)
{
  const int wave = threadIdx.x >> 5;
  const int wm = wave & 3, wn = wave >> 2;
  const size_t rowBase = (size_t)blockIdx.y * 64 + (size_t)wm * 16;
  const size_t colBase = (size_t)blockIdx.x * 128 + (size_t)wn * 64;

  v8f c0, c1, c2, c3;
  wave_gemm_16x64_acc(A + rowBase * (size_t)lda, (size_t)lda,
                      W + colBase * (size_t)ldw, (size_t)ldw,
                      bias + colBase, K, c0, c1, c2, c3);

  const int lane = threadIdx.x & 31;
  const int half = lane >> 4, l15 = lane & 15;
  const int rbase = half * 8;
  _Float16* o = out + rowBase * (size_t)ldo + colBase;
#pragma unroll
  for (int j = 0; j < 8; ++j) {
    o[(size_t)(rbase + j) * ldo +      l15] = (_Float16)c0[j];
    o[(size_t)(rbase + j) * ldo + 16 + l15] = (_Float16)c1[j];
    o[(size_t)(rbase + j) * ldo + 32 + l15] = (_Float16)c2[j];
    o[(size_t)(rbase + j) * ldo + 48 + l15] = (_Float16)c3[j];
  }
}

// ---------------------------------------------------------------------------
// Fused GRU step: one launch per timestep.
//   hg = hin @ Whh^T + bhh  (WMMA, staged in LDS)
//   h  = (1-z)*n + z*hin    (gate phase, same block via __syncthreads)
// Block = 384 threads = 12 waves: wave = (gate-chunk 0..2) x (M-group 0..3).
// Block covers 64 rows x 64 hidden-cols x all 3 gate chunks.
// grid = (HH/64 = 8, BB/64 = 2, ndir). Hidden state is ping-pong buffered
// (hin/hout) so unsynchronized blocks never race on h.
// ---------------------------------------------------------------------------
__device__ __forceinline__ float sigf(float x) { return 1.0f / (1.0f + __expf(-x)); }

__global__ __launch_bounds__(384) void gru_fused_step(
    const _Float16* __restrict__ hinF, _Float16* __restrict__ houtF,
    const _Float16* __restrict__ WF, const float* __restrict__ bF,
    const _Float16* __restrict__ xgF, _Float16* __restrict__ oF,
    const _Float16* __restrict__ hinB, _Float16* __restrict__ houtB,
    const _Float16* __restrict__ WB, const float* __restrict__ bB,
    const _Float16* __restrict__ xgB, _Float16* __restrict__ oB,
    long xg_bstride, long out_bstride)
{
  __shared__ float sm[3 * 64 * LDST];   // [chunk][row 0..63][col 0..63] padded

  const _Float16* hin  = blockIdx.z ? hinB  : hinF;
  _Float16*       hout = blockIdx.z ? houtB : houtF;
  const _Float16* W    = blockIdx.z ? WB    : WF;
  const float*    bb   = blockIdx.z ? bB    : bF;
  const _Float16* xg   = blockIdx.z ? xgB   : xgF;
  _Float16*       o    = blockIdx.z ? oB    : oF;

  const int jg    = blockIdx.x * 64;    // hidden-column group (0..511)
  const int mhalf = blockIdx.y * 64;    // row half (0 or 64)

  const int wave  = threadIdx.x >> 5;   // 0..11
  const int chunk = wave >> 2;          // 0..2 -> r/z/n gate chunk
  const int wm    = wave & 3;           // 0..3 -> 16-row group

  // --- WMMA phase: hg tile -> LDS ---
  {
    const size_t rowBase = (size_t)mhalf + (size_t)wm * 16;
    const size_t colBase = (size_t)chunk * HH + (size_t)jg;
    v8f c0, c1, c2, c3;
    wave_gemm_16x64_acc(hin + rowBase * (size_t)HH, (size_t)HH,
                        W + colBase * (size_t)HH, (size_t)HH,
                        bb + colBase, HH, c0, c1, c2, c3);

    const int lane = threadIdx.x & 31;
    const int half = lane >> 4, l15 = lane & 15;
    float* s = sm + chunk * 64 * LDST + (wm * 16 + half * 8) * LDST;
#pragma unroll
    for (int j = 0; j < 8; ++j) {
      s[j * LDST +      l15] = c0[j];
      s[j * LDST + 16 + l15] = c1[j];
      s[j * LDST + 32 + l15] = c2[j];
      s[j * LDST + 48 + l15] = c3[j];
    }
  }
  __syncthreads();

  // --- gate phase: 64x64 elements, 384 threads ---
  for (int idx = threadIdx.x; idx < 64 * 64; idx += 384) {
    const int row = idx >> 6;            // 0..63
    const int col = idx & 63;            // 0..63
    const int b = mhalf + row;
    const int j = jg + col;

    const float rr = sm[0 * 64 * LDST + row * LDST + col];
    const float zz = sm[1 * 64 * LDST + row * LDST + col];
    const float nn = sm[2 * 64 * LDST + row * LDST + col];

    const _Float16* xr = xg + (size_t)b * xg_bstride;
    float r = sigf((float)xr[j]          + rr);
    float z = sigf((float)xr[512 + j]    + zz);
    float n = tanhf((float)xr[1024 + j]  + r * nn);
    float hp = (float)hin[(size_t)b * HH + j];
    float hn = (1.0f - z) * n + z * hp;
    hout[(size_t)b * HH + j] = (_Float16)hn;
    if (o) o[(size_t)b * out_bstride + j] = (_Float16)hn;
  }
}

// Layer-1 backward contributes only its FIRST step (t=T-1, h0=0): hg == bhh.
__global__ __launch_bounds__(256) void gru_bwd_first_step(
    const _Float16* __restrict__ xg, const float* __restrict__ bhh,
    float* __restrict__ h2b)
{
  const int idx = blockIdx.x * blockDim.x + threadIdx.x;
  const int b = idx >> 9, j = idx & 511;
  const _Float16* xr = xg + (size_t)b * GG;
  float r = sigf((float)xr[j]        + bhh[j]);
  float z = sigf((float)xr[512 + j]  + bhh[512 + j]);
  float n = tanhf((float)xr[1024 + j] + r * bhh[1024 + j]);
  h2b[(size_t)b * HH + j] = (1.0f - z) * n;   // + z*0
}

// ---------------------------------------------------------------------------
// Small helper kernels
// ---------------------------------------------------------------------------
__global__ void embed_kernel(const int* __restrict__ tokens,
                             const float* __restrict__ emb,
                             _Float16* __restrict__ A0)
{
  const int row = blockIdx.x;                 // b*T + t
  const int tok = tokens[row];
  for (int c = threadIdx.x; c < EP; c += blockDim.x)
    A0[(size_t)row * EP + c] = (c < EE) ? (_Float16)emb[(size_t)tok * EE + c]
                                        : (_Float16)0.0f;
}

__global__ void wconv_kernel(const float* __restrict__ src, _Float16* __restrict__ dst,
                             int N, int Ks, int Kd)
{
  const long i = (long)blockIdx.x * blockDim.x + threadIdx.x;
  if (i >= (long)N * Kd) return;
  const int n = (int)(i / Kd), k = (int)(i % Kd);
  dst[i] = (k < Ks) ? (_Float16)src[(size_t)n * Ks + k] : (_Float16)0.0f;
}

__global__ void zero_f16_kernel(_Float16* __restrict__ p, long n)
{
  const long i = (long)blockIdx.x * blockDim.x + threadIdx.x;
  if (i < n) p[i] = (_Float16)0.0f;
}

__global__ void fc_kernel(const _Float16* __restrict__ h2f, const float* __restrict__ h2b,
                          const float* __restrict__ fcw, const float* __restrict__ fcb,
                          float* __restrict__ out)
{
  const int idx = blockIdx.x * blockDim.x + threadIdx.x;
  if (idx >= BB * 8) return;
  const int b = idx >> 3, o = idx & 7;
  float acc = fcb[o];
  const float* wr = fcw + (size_t)o * (2 * HH);
  for (int j = 0; j < HH; ++j) acc += (float)h2f[(size_t)b * HH + j] * wr[j];
  for (int j = 0; j < HH; ++j) acc += h2b[(size_t)b * HH + j] * wr[HH + j];
  out[(size_t)b * 8 + o] = acc;
}

// ---------------------------------------------------------------------------
// Host orchestration
// ---------------------------------------------------------------------------
extern "C" void kernel_launch(void* const* d_in, const int* in_sizes, int n_in,
                              void* d_out, int out_size, void* d_ws, size_t ws_size,
                              hipStream_t stream)
{
  (void)in_sizes; (void)n_in; (void)out_size; (void)ws_size;

  const int*   tokens = (const int*)  d_in[0];
  const float* emb    = (const float*)d_in[1];
  const float* Wih0f  = (const float*)d_in[2];
  const float* Whh0f  = (const float*)d_in[3];
  const float* bih0f  = (const float*)d_in[4];
  const float* bhh0f  = (const float*)d_in[5];
  const float* Wih0b  = (const float*)d_in[6];
  const float* Whh0b  = (const float*)d_in[7];
  const float* bih0b  = (const float*)d_in[8];
  const float* bhh0b  = (const float*)d_in[9];
  const float* Wih1f  = (const float*)d_in[10];
  const float* Whh1f  = (const float*)d_in[11];
  const float* bih1f  = (const float*)d_in[12];
  const float* bhh1f  = (const float*)d_in[13];
  const float* Wih1b  = (const float*)d_in[14];
  /* Whh1b (d_in[15]) unused: backward layer-1 needs only step t=T-1 from h0=0 */
  const float* bih1b  = (const float*)d_in[16];
  const float* bhh1b  = (const float*)d_in[17];
  const float* fc_w   = (const float*)d_in[18];
  const float* fc_b   = (const float*)d_in[19];
  float* outp = (float*)d_out;

  // workspace carve-up (256B aligned)
  char* base = (char*)d_ws; size_t off = 0;
  auto alloc = [&](size_t bytes) -> void* {
    void* r = base + off;
    off += (bytes + 255) & ~(size_t)255;
    return r;
  };
  _Float16* wih0f16 = (_Float16*)alloc((size_t)GG * EP * 2);
  _Float16* wih0b16 = (_Float16*)alloc((size_t)GG * EP * 2);
  _Float16* whh0f16 = (_Float16*)alloc((size_t)GG * HH * 2);
  _Float16* whh0b16 = (_Float16*)alloc((size_t)GG * HH * 2);
  _Float16* whh1f16 = (_Float16*)alloc((size_t)GG * HH * 2);
  _Float16* wih1f16 = (_Float16*)alloc((size_t)GG * 2 * HH * 2);
  _Float16* wih1b16 = (_Float16*)alloc((size_t)GG * 2 * HH * 2);
  _Float16* A0      = (_Float16*)alloc((size_t)MM * EP * 2);
  _Float16* xg0f    = (_Float16*)alloc((size_t)MM * GG * 2);
  _Float16* xg0b    = (_Float16*)alloc((size_t)MM * GG * 2);
  _Float16* h1seq   = (_Float16*)alloc((size_t)MM * 2 * HH * 2);
  _Float16* xg1f    = (_Float16*)alloc((size_t)MM * GG * 2);
  _Float16* xg1b    = (_Float16*)alloc((size_t)BB * GG * 2);
  // ping-pong hidden-state buffers (A buffers zero-initialized below)
  _Float16* h0fA    = (_Float16*)alloc((size_t)BB * HH * 2);
  _Float16* h0bA    = (_Float16*)alloc((size_t)BB * HH * 2);
  _Float16* h0fB    = (_Float16*)alloc((size_t)BB * HH * 2);
  _Float16* h0bB    = (_Float16*)alloc((size_t)BB * HH * 2);
  _Float16* h1A     = (_Float16*)alloc((size_t)BB * HH * 2);
  _Float16* h1B     = (_Float16*)alloc((size_t)BB * HH * 2);
  float*    h2b     = (float*)   alloc((size_t)BB * HH * 4);

  // --- weight conversion f32 -> f16 (zero-padded K for layer-0 Wih) ---
  auto wc = [&](const float* s, _Float16* d, int N, int Ks, int Kd) {
    long n = (long)N * Kd;
    wconv_kernel<<<dim3((unsigned)((n + 255) / 256)), dim3(256), 0, stream>>>(s, d, N, Ks, Kd);
  };
  wc(Wih0f, wih0f16, GG, EE, EP);
  wc(Wih0b, wih0b16, GG, EE, EP);
  wc(Whh0f, whh0f16, GG, HH, HH);
  wc(Whh0b, whh0b16, GG, HH, HH);
  wc(Whh1f, whh1f16, GG, HH, HH);
  wc(Wih1f, wih1f16, GG, 2 * HH, 2 * HH);
  wc(Wih1b, wih1b16, GG, 2 * HH, 2 * HH);

  // --- embedding gather -> f16, padded to EP ---
  embed_kernel<<<dim3(MM), dim3(64), 0, stream>>>(tokens, emb, A0);

  // --- layer-0 input projections (batched over B*T) ---
  gemm_f16_kernel<<<dim3(GG / 128, MM / 64), dim3(256), 0, stream>>>(
      A0, EP, wih0f16, EP, bih0f, xg0f, GG, EP);
  gemm_f16_kernel<<<dim3(GG / 128, MM / 64), dim3(256), 0, stream>>>(
      A0, EP, wih0b16, EP, bih0b, xg0b, GG, EP);

  // --- zero initial hidden states (h0fA, h0bA contiguous) ---
  zero_f16_kernel<<<dim3(2 * BB * HH / 256), dim3(256), 0, stream>>>(h0fA, 2L * BB * HH);

  // --- layer-0 bidirectional scan: one fused launch per step ---
  {
    _Float16 *curF = h0fA, *nxtF = h0fB, *curB = h0bA, *nxtB = h0bB;
    for (int s = 0; s < TT; ++s) {
      gru_fused_step<<<dim3(HH / 64, BB / 64, 2), dim3(384), 0, stream>>>(
          curF, nxtF, whh0f16, bhh0f, xg0f + (size_t)s * GG,
          h1seq + (size_t)s * 2 * HH,
          curB, nxtB, whh0b16, bhh0b, xg0b + (size_t)(TT - 1 - s) * GG,
          h1seq + (size_t)(TT - 1 - s) * 2 * HH + HH,
          (long)TT * GG, (long)TT * 2 * HH);
      _Float16* t;
      t = curF; curF = nxtF; nxtF = t;
      t = curB; curB = nxtB; nxtB = t;
    }
  }

  // --- layer-1 input projections ---
  gemm_f16_kernel<<<dim3(GG / 128, MM / 64), dim3(256), 0, stream>>>(
      h1seq, 2 * HH, wih1f16, 2 * HH, bih1f, xg1f, GG, 2 * HH);
  // backward direction: only t = T-1 rows are needed (h1[:, T-1, :], row stride T*2H)
  gemm_f16_kernel<<<dim3(GG / 128, BB / 64), dim3(256), 0, stream>>>(
      h1seq + (size_t)(TT - 1) * 2 * HH, TT * 2 * HH, wih1b16, 2 * HH, bih1b, xg1b, GG, 2 * HH);

  // --- layer-1 forward scan (only final hidden needed -> no seq output) ---
  zero_f16_kernel<<<dim3(BB * HH / 256), dim3(256), 0, stream>>>(h1A, (long)BB * HH);
  _Float16* cur1 = h1A;
  {
    _Float16* nxt1 = h1B;
    for (int s = 0; s < TT; ++s) {
      gru_fused_step<<<dim3(HH / 64, BB / 64, 1), dim3(384), 0, stream>>>(
          cur1, nxt1, whh1f16, bhh1f, xg1f + (size_t)s * GG, nullptr,
          nullptr, nullptr, nullptr, nullptr, nullptr, nullptr,
          (long)TT * GG, 0L);
      _Float16* t = cur1; cur1 = nxt1; nxt1 = t;
    }
  }
  // after the loop, cur1 holds the final forward hidden state of layer 1

  // --- layer-1 backward: single step from h0 = 0 (hg == bhh, no GEMM) ---
  gru_bwd_first_step<<<dim3(BB * HH / 256), dim3(256), 0, stream>>>(xg1b, bhh1b, h2b);

  // --- final FC: out[b,o] = concat(h2f, h2b) . fc_w[o] + fc_b[o] ---
  fc_kernel<<<dim3((BB * 8 + 255) / 256), dim3(256), 0, stream>>>(
      cur1, h2b, fc_w, fc_b, outp);
}
